// SRGAT_27341761806358
// MI455X (gfx1250) — matrix operand order repaired and checked by reading, not compiled, not run on hardware
//
#include <hip/hip_runtime.h>
#include <hip/hip_bf16.h>

// ---------------------------------------------------------------------------
// SRGAT forward, CDNA5 (gfx1250) implementation.
// All GEMMs use V_WMMA_F32_16X16X32_BF16 (bf16 in, f32 accumulate).
// Pair-bias MLP runs on packed-f32 VALU (v_pk_fma_f32 path).
// N=1024 agents, H=64, P=2 passes, M=20 modes.
// ---------------------------------------------------------------------------

#define NAG 1024
#define HD  64
#define NMODE 20
#define LROW 1028   // padded LDS row stride (floats) for attention logits
#define PROW 1284   // padded LDS row stride (floats) for proj rows (1280+4)

typedef __bf16 bf16;
typedef __attribute__((ext_vector_type(16))) __bf16 bf16x16;
typedef __attribute__((ext_vector_type(8)))  float  f32x8;
typedef __attribute__((ext_vector_type(2)))  float  f32x2;

__device__ __forceinline__ f32x8 wmma_bf16(bf16x16 a, bf16x16 b, f32x8 c) {
  // D = A(16x32, bf16) * B(32x16, bf16) + C(16x16, f32)
  return __builtin_amdgcn_wmma_f32_16x16x32_bf16(false, a, false, b,
                                                 (short)0, c, false, false);
}

// A-operand 16x32 slice, source row-major f32 (global or LDS).
// rowptr points at this lane's A row (M = lane&15). Element e of lane l maps
// to K = kslice + half*16 + e  -> one contiguous 16-run.
__device__ __forceinline__ bf16x16 a_slice_f32(const float* rowptr, int kslice, int half) {
  const float* r0 = rowptr + kslice + half * 16;
  bf16x16 a;
#pragma unroll
  for (int e = 0; e < 8; ++e) a[e]     = (bf16)r0[e];
#pragma unroll
  for (int e = 0; e < 8; ++e) a[8 + e] = (bf16)r0[8 + e];
  return a;
}

// Same mapping, source already bf16.
__device__ __forceinline__ bf16x16 a_slice_bf16(const bf16* rowptr, int kslice, int half) {
  const bf16* r0 = rowptr + kslice + half * 16;
  bf16x16 a;
#pragma unroll
  for (int e = 0; e < 8; ++e) a[e]     = r0[e];
#pragma unroll
  for (int e = 0; e < 8; ++e) a[8 + e] = r0[8 + e];
  return a;
}

// B-operand 32x16 slice: element e of lane l maps to K = half*16 + e (contig),
// N = lane&15. Caller passes p = &B_colmajor[col*ldk + kslice + half*16].
__device__ __forceinline__ bf16x16 b_slice_bf16(const bf16* p) {
  bf16x16 b;
#pragma unroll
  for (int e = 0; e < 16; ++e) b[e] = p[e];
  return b;
}

__device__ __forceinline__ float sigmoidf_(float x) {
  return 1.0f / (1.0f + __expf(-x));
}

// ---------------------------------------------------------------------------
// Weight transpose + f32->bf16 convert: src[K][C] row-major -> dst[c*K + k]
// (column-major), so WMMA B-loads are contiguous over K.
// ---------------------------------------------------------------------------
__global__ void convT_kernel(const float* __restrict__ src, bf16* __restrict__ dst,
                             int K, int C) {
  int idx = blockIdx.x * 256 + threadIdx.x;
  if (idx >= K * C) return;
  int k = idx / C, c = idx % C;
  dst[c * K + k] = (bf16)src[idx];
}

// ---------------------------------------------------------------------------
// q,k,v projections.  768 waves: tile = rowtile(64) x {q,k,v}(3) x coltile(4).
// q is pre-scaled by 1/sqrt(H) and stored bf16 row-major [N][64];
// k stored bf16 row-major [N][64]; v stored bf16 TRANSPOSED [64][N].
// Store epilogue is branch-free: one uniform (base, stride, scale) triple.
// ---------------------------------------------------------------------------
__global__ __launch_bounds__(128) void qkv_kernel(
    const float* __restrict__ hin,
    const bf16* __restrict__ wqT, const bf16* __restrict__ wkT, const bf16* __restrict__ wvT,
    bf16* __restrict__ qbf, bf16* __restrict__ kbf, bf16* __restrict__ vbfT) {
  const int tid = threadIdx.x, wave = tid >> 5, lane = tid & 31;
  const int half = lane >> 4, l15 = lane & 15;
  const int tile = blockIdx.x * 4 + wave;       // 0..767
  const int rowt = tile / 12, rem = tile % 12;
  const int mat = rem >> 2, ct = rem & 3;
  const bf16* WT = (mat == 0) ? wqT : ((mat == 1) ? wkT : wvT);
  const float* arow = hin + (rowt * 16 + l15) * HD;
  f32x8 acc = {};
#pragma unroll
  for (int kk = 0; kk < 2; ++kk) {
    bf16x16 a = a_slice_f32(arow, kk * 32, half);
    bf16x16 b = b_slice_bf16(WT + (ct * 16 + l15) * HD + kk * 32 + half * 16);
    acc = wmma_bf16(a, b, acc);
  }
  const int c = ct * 16 + l15;
  // uniform scatter descriptor: q/k row-major (stride HD), vT col-major (stride 1)
  const float scale = (mat == 0) ? 0.125f : 1.0f;           // 1/sqrt(64) for q
  bf16* base = (mat == 0) ? (qbf + c) : ((mat == 1) ? (kbf + c)
                                                    : (vbfT + (size_t)c * NAG));
  const int stride = (mat == 2) ? 1 : HD;
  const int n0 = rowt * 16 + half * 8;
#pragma unroll
  for (int r = 0; r < 8; ++r)
    base[(size_t)(n0 + r) * stride] = (bf16)(acc[r] * scale);
}

// ---------------------------------------------------------------------------
// Fused attention pass: per block = one 16-row i-tile.
//  phase 1: bias-MLP (packed-f32 VALU, in C layout) + q@k^T (WMMA) + mask -> LDS
//  phase 2: row softmax (shfl reductions), zero rows with no neighbors
//  phase 3: attn@v (WMMA, split-K over 8 waves) -> msg[N][64] f32
// Dynamic LDS: 16*LROW floats (logits) + 8*256 floats (split-K partials).
// ---------------------------------------------------------------------------
__global__ __launch_bounds__(256) void attn_kernel(
    const bf16* __restrict__ qbf, const bf16* __restrict__ kbf,
    const bf16* __restrict__ vbfT,
    const float* __restrict__ positions,
    const float* __restrict__ w1,   // corr_w1[p] : [2][64]
    const float* __restrict__ b1,   // corr_b1[p] : [64]
    const float* __restrict__ w2,   // corr_w2[p] : [64]
    const unsigned char* __restrict__ mask,
    float* __restrict__ msg) {
  extern __shared__ float smem[];
  float* lds  = smem;                 // [16][LROW]
  float* part = smem + 16 * LROW;     // [8][16][16]

  const int tid = threadIdx.x, wave = tid >> 5, lane = tid & 31;
  const int half = lane >> 4, l15 = lane & 15;
  const int i0 = blockIdx.x * 16;
  const float* w1x = w1;
  const float* w1y = w1 + HD;

  float pxi[8], pyi[8];
#pragma unroll
  for (int r = 0; r < 8; ++r) {
    int i = i0 + half * 8 + r;
    pxi[r] = positions[2 * i];
    pyi[r] = positions[2 * i + 1];
  }
  const bf16* arow = qbf + (i0 + l15) * HD;   // A rows: M = lane&15

  // ---- phase 1: logits = scale*q@k^T + bias(i,j), masked ----
  for (int jt = wave; jt < NAG / 16; jt += 8) {
    const int j = jt * 16 + l15;
    // prefetch next j-tile's mask bytes for this i-tile (global_prefetch_b8)
    if (jt + 8 < NAG / 16)
      __builtin_prefetch(mask + (size_t)(i0 + half * 8) * NAG + (jt + 8) * 16 + l15, 0, 1);

    const float pxj = positions[2 * j], pyj = positions[2 * j + 1];
    // row-pair packed layout: encourages v_pk_fma_f32
    f32x2 dx2[4], dy2[4], bacc[4];
#pragma unroll
    for (int rp = 0; rp < 4; ++rp) {
      dx2[rp] = (f32x2){pxi[2 * rp] - pxj, pxi[2 * rp + 1] - pxj};
      dy2[rp] = (f32x2){pyi[2 * rp] - pyj, pyi[2 * rp + 1] - pyj};
      bacc[rp] = (f32x2){0.0f, 0.0f};
    }
    // pair-bias MLP: bias = sum_h relu(dx*w1x[h] + dy*w1y[h] + b1[h]) * w2[h]
    for (int hh = 0; hh < HD; ++hh) {
      const f32x2 axv = {w1x[hh], w1x[hh]};
      const f32x2 ayv = {w1y[hh], w1y[hh]};
      const f32x2 bbv = {b1[hh],  b1[hh]};
      const f32x2 wwv = {w2[hh],  w2[hh]};
#pragma unroll
      for (int rp = 0; rp < 4; ++rp) {
        f32x2 t = __builtin_elementwise_fma(dx2[rp], axv,
                   __builtin_elementwise_fma(dy2[rp], ayv, bbv));
        t = __builtin_elementwise_max(t, (f32x2){0.0f, 0.0f});
        bacc[rp] = __builtin_elementwise_fma(t, wwv, bacc[rp]);
      }
    }
    f32x8 c;
#pragma unroll
    for (int r = 0; r < 8; ++r) c[r] = bacc[r >> 1][r & 1];

    // + q (pre-scaled) @ k^T over K=64
#pragma unroll
    for (int kk = 0; kk < 2; ++kk) {
      bf16x16 a = a_slice_bf16(arow, kk * 32, half);
      bf16x16 b = b_slice_bf16(kbf + j * HD + kk * 32 + half * 16);
      c = wmma_bf16(a, b, c);
    }
    // mask & stash row block in LDS
#pragma unroll
    for (int r = 0; r < 8; ++r) {
      int i = i0 + half * 8 + r;
      float v = mask[(size_t)i * NAG + j] ? c[r] : -1e9f;
      lds[(half * 8 + r) * LROW + jt * 16 + l15] = v;
    }
  }
  __syncthreads();

  // ---- phase 2: softmax per row (16 threads/row) ----
  {
    const int row = tid >> 4, cs = tid & 15;
    float mx = -3.4e38f;
    for (int jj = cs; jj < NAG; jj += 16) mx = fmaxf(mx, lds[row * LROW + jj]);
#pragma unroll
    for (int m = 1; m < 16; m <<= 1) mx = fmaxf(mx, __shfl_xor(mx, m, 32));
    const bool has = (mx > -1e8f);   // all-masked rows have mx == -1e9
    float sum = 0.0f;
    for (int jj = cs; jj < NAG; jj += 16) {
      float e_ = __expf(lds[row * LROW + jj] - mx);
      lds[row * LROW + jj] = e_;
      sum += e_;
    }
#pragma unroll
    for (int m = 1; m < 16; m <<= 1) sum += __shfl_xor(sum, m, 32);
    const float inv = has ? (1.0f / sum) : 0.0f;  // msg = 0 when no neighbors
    for (int jj = cs; jj < NAG; jj += 16) lds[row * LROW + jj] *= inv;
  }
  __syncthreads();

  // ---- phase 3: msg = attn @ v (split-K: wave = coltile(4) x khalf(2)) ----
  {
    const int ct = wave & 3, kh = wave >> 2;
    const float* arow2 = lds + l15 * LROW;   // A rows: M = lane&15
    f32x8 acc = {};
    for (int s = 0; s < 16; ++s) {
      const int kslice = kh * 512 + s * 32;
      bf16x16 a = a_slice_f32(arow2, kslice, half);
      bf16x16 b = b_slice_bf16(vbfT + (ct * 16 + l15) * NAG + kslice + half * 16);
      acc = wmma_bf16(a, b, acc);
    }
    float* pp = part + wave * 256;
#pragma unroll
    for (int r = 0; r < 8; ++r) pp[(half * 8 + r) * 16 + l15] = acc[r];
  }
  __syncthreads();
  for (int o = tid; o < 16 * HD; o += 256) {
    const int il = o >> 6, c = o & 63;
    const int ct = c >> 4, cl = c & 15;
    msg[(i0 + il) * HD + c] =
        part[ct * 256 + il * 16 + cl] + part[(4 + ct) * 256 + il * 16 + cl];
  }
}

// ---------------------------------------------------------------------------
// Sigmoid-gated combine: g = sigmoid([A1,A2] @ W + b)
//   selA=0 (message gate): out = (1-g)*A1 + g*A2     (A1=h, A2=msg)
//   selA=1 (fuse):         out = g*A1 + (1-g)*A2     (A1=x_enc, A2=h)
// WT is [64 cols][128 K] column-major bf16.
// ---------------------------------------------------------------------------
__global__ __launch_bounds__(128) void gatefuse_kernel(
    const float* __restrict__ A1, const float* __restrict__ A2,
    const bf16* __restrict__ WT, const float* __restrict__ bias,
    float* __restrict__ out, int selA) {
  const int tid = threadIdx.x, wave = tid >> 5, lane = tid & 31;
  const int half = lane >> 4, l15 = lane & 15;
  const int tile = blockIdx.x * 4 + wave;        // 0..255
  const int rowt = tile >> 2, ct = tile & 3;
  const float* a1row = A1 + (rowt * 16 + l15) * HD;
  const float* a2row = A2 + (rowt * 16 + l15) * HD;
  f32x8 acc = {};
#pragma unroll
  for (int kk = 0; kk < 4; ++kk) {
    const float* src = (kk < 2) ? a1row : a2row;
    bf16x16 a = a_slice_f32(src, (kk & 1) * 32, half);
    bf16x16 b = b_slice_bf16(WT + (ct * 16 + l15) * 128 + kk * 32 + half * 16);
    acc = wmma_bf16(a, b, acc);
  }
  const int c = ct * 16 + l15;
  const float bb = bias[c];
#pragma unroll
  for (int r = 0; r < 8; ++r) {
    const int n = rowt * 16 + half * 8 + r;
    const float g = sigmoidf_(acc[r] + bb);
    const float ga = selA ? g : (1.0f - g);      // coefficient on A1
    const float v1 = A1[n * HD + c], v2 = A2[n * HD + c];
    out[n * HD + c] = ga * v1 + (1.0f - ga) * v2;
  }
}

// ---------------------------------------------------------------------------
// Fused head: proj (64->1280) + bias -> LayerNorm -> ReLU  (all in LDS)
//             -> dest_w1 (64->64) + b1 -> dest_w2 (64->2) + b2 -> out[M][N][2]
// One block per 16 agents. Dynamic LDS: 16*PROW floats.
// ---------------------------------------------------------------------------
__global__ __launch_bounds__(256) void head_kernel(
    const float* __restrict__ fused,
    const bf16* __restrict__ projT, const float* __restrict__ proj_b,
    const float* __restrict__ ln_g, const float* __restrict__ ln_b,
    const bf16* __restrict__ destT, const float* __restrict__ dest_b1,
    const float* __restrict__ dest_w2, const float* __restrict__ dest_b2,
    float* __restrict__ out) {
  extern __shared__ float smem[];
  float* lds = smem;   // [16][PROW]
  const int tid = threadIdx.x, wave = tid >> 5, lane = tid & 31;
  const int half = lane >> 4, l15 = lane & 15;
  const int n0 = blockIdx.x * 16;

  // ---- phase A: proj tile into LDS (80 col tiles over 8 waves) ----
  const float* arow = fused + (n0 + l15) * HD;
#pragma unroll
  for (int t = 0; t < 10; ++t) {
    const int ct = wave + 8 * t;
    f32x8 acc = {};
#pragma unroll
    for (int kk = 0; kk < 2; ++kk) {
      bf16x16 a = a_slice_f32(arow, kk * 32, half);
      bf16x16 b = b_slice_bf16(projT + (ct * 16 + l15) * HD + kk * 32 + half * 16);
      acc = wmma_bf16(a, b, acc);
    }
    const int c = ct * 16 + l15;
    const float pb = proj_b[c];
#pragma unroll
    for (int r = 0; r < 8; ++r) lds[(half * 8 + r) * PROW + c] = acc[r] + pb;
  }
  __syncthreads();

  // ---- phase B: LayerNorm + ReLU per row (16 threads/row over 1280) ----
  {
    const int row = tid >> 4, cs = tid & 15;
    float s = 0.0f, s2 = 0.0f;
    for (int c = cs; c < NMODE * HD; c += 16) {
      float x = lds[row * PROW + c];
      s += x;
      s2 = fmaf(x, x, s2);
    }
#pragma unroll
    for (int m = 1; m < 16; m <<= 1) {
      s  += __shfl_xor(s,  m, 32);
      s2 += __shfl_xor(s2, m, 32);
    }
    const float mu  = s * (1.0f / (NMODE * HD));
    const float var = s2 * (1.0f / (NMODE * HD)) - mu * mu;
    const float rs  = rsqrtf(var + 1e-5f);
    for (int c = cs; c < NMODE * HD; c += 16) {
      float x = lds[row * PROW + c];
      float y = (x - mu) * rs * ln_g[c] + ln_b[c];
      lds[row * PROW + c] = fmaxf(y, 0.0f);
    }
  }
  __syncthreads();

  // ---- phase C: dest. 320 local rows (16 agents x 20 modes) = 20 row tiles.
  //      dest_w2 (64->2) folded into a cross-lane-reduced epilogue. ----
  float w20[4], w21[4], b1v[4];
#pragma unroll
  for (int ct = 0; ct < 4; ++ct) {
    const int hc = ct * 16 + l15;
    w20[ct] = dest_w2[hc * 2 + 0];
    w21[ct] = dest_w2[hc * 2 + 1];
    b1v[ct] = dest_b1[hc];
  }
  const float db20 = dest_b2[0], db21 = dest_b2[1];

  for (int rt = wave; rt < 20; rt += 8) {
    const int rg = rt * 16 + l15;             // A row (local): n_loc*20 + mode
    const float* arow2 = lds + (rg / NMODE) * PROW + (rg % NMODE) * HD;
    f32x8 acc[4] = {{}, {}, {}, {}};
#pragma unroll
    for (int kk = 0; kk < 2; ++kk) {
      bf16x16 a = a_slice_f32(arow2, kk * 32, half);
#pragma unroll
      for (int ct = 0; ct < 4; ++ct) {
        bf16x16 b = b_slice_bf16(destT + (ct * 16 + l15) * HD + kk * 32 + half * 16);
        acc[ct] = wmma_bf16(a, b, acc[ct]);
      }
    }
#pragma unroll
    for (int r = 0; r < 8; ++r) {
      const int rg2 = rt * 16 + half * 8 + r;   // output row (local)
      const int n = n0 + rg2 / NMODE, md = rg2 % NMODE;
      float p0 = 0.0f, p1 = 0.0f;
#pragma unroll
      for (int ct = 0; ct < 4; ++ct) {
        const float tv = acc[ct][r] + b1v[ct];
        p0 = fmaf(tv, w20[ct], p0);
        p1 = fmaf(tv, w21[ct], p1);
      }
#pragma unroll
      for (int m = 1; m < 16; m <<= 1) {
        p0 += __shfl_xor(p0, m, 32);
        p1 += __shfl_xor(p1, m, 32);
      }
      if (l15 == 0) {
        out[(md * NAG + n) * 2 + 0] = p0 + db20;
        out[(md * NAG + n) * 2 + 1] = p1 + db21;
      }
    }
  }
}

// ---------------------------------------------------------------------------
// Host launcher
// ---------------------------------------------------------------------------
extern "C" void kernel_launch(void* const* d_in, const int* in_sizes, int n_in,
                              void* d_out, int out_size, void* d_ws, size_t ws_size,
                              hipStream_t stream) {
  const float* x_enc = (const float*)d_in[0];
  const float* h0    = (const float*)d_in[1];
  const float* pos   = (const float*)d_in[2];
  const float* Wq    = (const float*)d_in[3];
  const float* Wk    = (const float*)d_in[4];
  const float* Wv    = (const float*)d_in[5];
  const float* cw1   = (const float*)d_in[6];
  const float* cb1   = (const float*)d_in[7];
  const float* cw2   = (const float*)d_in[8];
  const float* gw    = (const float*)d_in[9];
  const float* gb    = (const float*)d_in[10];
  const float* fw    = (const float*)d_in[11];
  const float* fb    = (const float*)d_in[12];
  const float* pw    = (const float*)d_in[13];
  const float* pb    = (const float*)d_in[14];
  const float* lng   = (const float*)d_in[15];
  const float* lnb   = (const float*)d_in[16];
  const float* dw1   = (const float*)d_in[17];
  const float* db1   = (const float*)d_in[18];
  const float* dw2   = (const float*)d_in[19];
  const float* db2   = (const float*)d_in[20];
  const unsigned char* mask = (const unsigned char*)d_in[21];
  float* out = (float*)d_out;

  // Workspace carve (256B aligned)
  char* base = (char*)d_ws;
  size_t off = 0;
  auto carve = [&](size_t bytes) -> char* {
    char* p = base + off;
    off = (off + bytes + 255) & ~(size_t)255;
    return p;
  };
  bf16* wqT   = (bf16*)carve(2 * 64 * 64 * sizeof(bf16));     // per pass
  bf16* wkT   = (bf16*)carve(2 * 64 * 64 * sizeof(bf16));
  bf16* wvT   = (bf16*)carve(2 * 64 * 64 * sizeof(bf16));
  bf16* gwT   = (bf16*)carve(2 * 128 * 64 * sizeof(bf16));
  bf16* fwT   = (bf16*)carve(128 * 64 * sizeof(bf16));
  bf16* pwT   = (bf16*)carve(64 * 1280 * sizeof(bf16));
  bf16* dwT   = (bf16*)carve(64 * 64 * sizeof(bf16));
  bf16* qbf   = (bf16*)carve((size_t)NAG * HD * sizeof(bf16));
  bf16* kbf   = (bf16*)carve((size_t)NAG * HD * sizeof(bf16));
  bf16* vbfT  = (bf16*)carve((size_t)NAG * HD * sizeof(bf16));
  float* msg  = (float*)carve((size_t)NAG * HD * sizeof(float));
  float* hA   = (float*)carve((size_t)NAG * HD * sizeof(float));
  float* hB   = (float*)carve((size_t)NAG * HD * sizeof(float));
  float* fus  = (float*)carve((size_t)NAG * HD * sizeof(float));

  // ---- prep: transpose+convert weights to bf16 column-major ----
  auto convT = [&](const float* src, bf16* dst, int K, int C) {
    int tot = K * C;
    convT_kernel<<<(tot + 255) / 256, 256, 0, stream>>>(src, dst, K, C);
  };
  for (int p = 0; p < 2; ++p) {
    convT(Wq + p * 4096, wqT + p * 4096, 64, 64);
    convT(Wk + p * 4096, wkT + p * 4096, 64, 64);
    convT(Wv + p * 4096, wvT + p * 4096, 64, 64);
    convT(gw + p * 8192, gwT + p * 8192, 128, 64);
  }
  convT(fw, fwT, 128, 64);
  convT(pw, pwT, 64, 1280);
  convT(dw1, dwT, 64, 64);

  const size_t attn_lds = (size_t)(16 * LROW + 8 * 256) * sizeof(float);
  const size_t head_lds = (size_t)(16 * PROW) * sizeof(float);

  // ---- message-passing passes ----
  const float* h_in = h0;
  float* h_out = hA;
  for (int p = 0; p < 2; ++p) {
    qkv_kernel<<<192, 128, 0, stream>>>(h_in,
                                        wqT + p * 4096, wkT + p * 4096, wvT + p * 4096,
                                        qbf, kbf, vbfT);
    attn_kernel<<<64, 256, attn_lds, stream>>>(qbf, kbf, vbfT, pos,
                                               cw1 + p * 128, cb1 + p * 64,
                                               cw2 + p * 64, mask, msg);
    gatefuse_kernel<<<64, 128, 0, stream>>>(h_in, msg, gwT + p * 8192,
                                            gb + p * 64, h_out, /*selA=*/0);
    h_in = h_out;
    h_out = hB;
  }

  // ---- fuse with temporal encoding ----
  gatefuse_kernel<<<64, 128, 0, stream>>>(x_enc, h_in, fwT, fb, fus, /*selA=*/1);

  // ---- fused proj + LN + ReLU + decoder ----
  head_kernel<<<64, 256, head_lds, stream>>>(fus, pwT, pb, lng, lnb,
                                             dwT, db1, dw2, db2, out);
}
